// ButterflyModule_71176198029535
// MI455X (gfx1250) — matrix operand order repaired and checked by reading, not compiled, not run on hardware
//
#include <hip/hip_runtime.h>

// ---------------------------------------------------------------------------
// Butterfly(W=256, 16 layers) == linear map: out = data x G  (G: 256x256).
// MI455X roofline: 256 MB traffic / 23.3 TB/s ~= 11us floor; bf16 WMMA math
// (3-term hi/lo split, fp32 accumulate ~= fp32 accuracy) hides entirely under
// the memory floor. G is composed on-device from the exact fp32 butterfly
// applied to the identity (gather/scatter indices folded in for free).
// ---------------------------------------------------------------------------

#define WDIM 256
#define GELEMS (WDIM * WDIM)      // 65536
#define LDS_BF GELEMS             // bf16 elements per (hi|lo) half

typedef __attribute__((ext_vector_type(16))) __bf16 v16bf;
typedef __attribute__((ext_vector_type(8)))  float  v8f;
typedef __attribute__((ext_vector_type(4)))  int    v4i;

typedef __attribute__((address_space(1))) v4i* as1_v4i_p;
typedef __attribute__((address_space(3))) v4i* as3_v4i_p;

// ---------------------------------------------------------------------------
// Kernel 1: compose G = full_op(identity).  One row per wave32 (8 f32/lane).
// Strides 1,2,4 stay inside a lane's 8 elements; strides 8..128 are lane-XOR
// exchanges with mask s/8 (all within wave32).
// ---------------------------------------------------------------------------
__global__ __launch_bounds__(256) void compose_kernel(
    const float* __restrict__ angles,   // (16,128)
    const int*   __restrict__ idx_in,   // (256,)
    const int*   __restrict__ idx_out,  // (256,)
    float*       __restrict__ G)        // (256,256) out
{
  const int lane = threadIdx.x & 31;
  const int row  = blockIdx.x * 8 + (threadIdx.x >> 5);

  float x[8];
#pragma unroll
  for (int j = 0; j < 8; ++j) {
    const int w = lane * 8 + j;
    x[j] = (idx_in[w] == row) ? 1.0f : 0.0f;   // x = I[:, idx_in]
  }

  const int strides[16] = {1,2,4,8,16,32,64,128,128,64,32,16,8,4,2,1};
#pragma unroll
  for (int l = 0; l < 16; ++l) {
    const int s  = strides[l];
    const int ls = 31 - __clz(s);
    const float* al = angles + l * 128;
    if (s < 8) {
      // in-lane pairs (j, j+s)
#pragma unroll
      for (int j = 0; j < 8; ++j) {
        if (((j >> ls) & 1) == 0) {
          const int jj  = j + s;
          const int w   = lane * 8 + j;
          const int idx = ((w >> (ls + 1)) << ls) | (w & (s - 1));
          float sn, c; sincosf(al[idx], &sn, &c);
          const float a = x[j], b = x[jj];
          x[j]  = a * c  - b * sn;
          x[jj] = a * sn + b * c;
        }
      }
    } else {
      const int mask = s >> 3;   // partner lane = lane ^ mask
#pragma unroll
      for (int j = 0; j < 8; ++j) {
        const int w   = lane * 8 + j;
        const int idx = ((w >> (ls + 1)) << ls) | (w & (s - 1));
        float sn, c; sincosf(al[idx], &sn, &c);
        const float p = __shfl_xor(x[j], mask, 32);
        if (((w >> ls) & 1) == 0) x[j] = x[j] * c + p * (-sn);
        else                      x[j] = p * sn + x[j] * c;
      }
    }
  }

#pragma unroll
  for (int j = 0; j < 8; ++j) {
    const int w = lane * 8 + j;
    G[row * WDIM + idx_out[w]] = x[j];   // out[:, idx_out] = y
  }
}

// ---------------------------------------------------------------------------
// Kernel 2: split G into bf16 hi/lo, pre-swizzled into WMMA B-fragment layout.
// B (32x16 per tile), contiguous k order chosen here: lanes 0-15 hold K=0..15, lanes 16-31 K=16..31,
// slot j within a lane's 16 bf16 <-> K = 16*(lane>>4) + j;  N = lane & 15.
// Fragment (nt,kt) stored at ((nt*8+kt)*1024 + lane*16 + slot).
// ---------------------------------------------------------------------------
__global__ __launch_bounds__(256) void split_kernel(
    const float* __restrict__ G,
    __bf16*      __restrict__ bhi,
    __bf16*      __restrict__ blo)
{
  const int i = blockIdx.x * 256 + threadIdx.x;   // 0..65535
  const int k = i >> 8;        // K index (row of G)
  const int n = i & 255;       // N index (col of G)
  const float g = G[i];
  const __bf16 h  = (__bf16)g;
  const __bf16 lo = (__bf16)(g - (float)h);
  const int kt = k >> 5, ntile = n >> 4;
  const int kl = k & 31, nl   = n & 15;
  const int lanef = nl + ((kl >> 4) << 4);
  const int slot  = kl & 15;
  const int off   = (ntile * 8 + kt) * 1024 + lanef * 16 + slot;
  bhi[off] = h;
  blo[off] = lo;
}

// ---------------------------------------------------------------------------
// Kernel 3: out(131072x256,f32) = data x G via v_wmma_f32_16x16x32_bf16.
// 512 threads (16 waves) per block, 256 rows/block, 512 blocks.
// B hi+lo (256 KB) staged in LDS (320 KB WGP LDS) via CDNA5 async-to-LDS.
// Each wave keeps its 16x256 A panel (hi+lo bf16 fragments, 128 VGPRs)
// resident and sweeps 16 N-tiles x 8 K-tiles with 3 WMMAs per K-step.
// ---------------------------------------------------------------------------
__global__ __launch_bounds__(512) void wmma_gemm_kernel(
    const float*  __restrict__ data,   // (131072, 256)
    const __bf16* __restrict__ bfr,    // bhi (65536) immediately followed by blo
    float*        __restrict__ out)    // (131072, 256)
{
  __shared__ __bf16 lds[2 * LDS_BF];   // 256 KB: [hi | lo]

  // ---- stage B fragments into LDS: 262144 B = 16384 x 16B chunks ----
#if __has_builtin(__builtin_amdgcn_global_load_async_to_lds_b128)
  for (int t = threadIdx.x; t < 16384; t += 512) {
    __builtin_amdgcn_global_load_async_to_lds_b128(
        (as1_v4i_p)((const char*)bfr + t * 16),
        (as3_v4i_p)((char*)lds + t * 16),
        0, 0);
  }
#if __has_builtin(__builtin_amdgcn_s_wait_asynccnt)
  __builtin_amdgcn_s_wait_asynccnt(0);
#else
  asm volatile("s_wait_asynccnt 0" ::: "memory");
#endif
#else
  for (int t = threadIdx.x; t < 16384; t += 512) {
    ((float4*)lds)[t] = ((const float4*)bfr)[t];
  }
#endif
  __syncthreads();

  const int lane    = threadIdx.x & 31;
  const int wave    = threadIdx.x >> 5;
  const int rowbase = blockIdx.x * 256 + wave * 16;
  const int r       = rowbase + (lane & 15);
  const int khalf   = (lane >> 4) << 4;   // 0 or 16: K-half owned by this lane
  const float* rowp = data + (size_t)r * WDIM;

  // ---- load A panel (16 rows x 256 K) as bf16 hi/lo fragments ----
  // A 16x32 fragment, contiguous k order (same as B): lane&15 = M; slots 0..15 -> K = khalf+0..15.
  v16bf ahi[8], alo[8];
#pragma unroll
  for (int kt = 0; kt < 8; ++kt) {
    const float* p = rowp + kt * 32 + khalf;
    const float4 f0 = *(const float4*)(p);
    const float4 f1 = *(const float4*)(p + 4);
    const float4 f2 = *(const float4*)(p + 8);
    const float4 f3 = *(const float4*)(p + 12);
    const float f[16] = {f0.x,f0.y,f0.z,f0.w, f1.x,f1.y,f1.z,f1.w,
                         f2.x,f2.y,f2.z,f2.w, f3.x,f3.y,f3.z,f3.w};
#pragma unroll
    for (int j = 0; j < 16; ++j) {
      const __bf16 h = (__bf16)f[j];
      ahi[kt][j] = h;
      alo[kt][j] = (__bf16)(f[j] - (float)h);
    }
  }

  // ---- GEMM sweep: 16 N-tiles, 8 K-steps, 3-term split per step ----
#pragma unroll 1
  for (int nt = 0; nt < 16; ++nt) {
    v8f acc = {};
#pragma unroll
    for (int kt = 0; kt < 8; ++kt) {
      const __bf16* bp = lds + (nt * 8 + kt) * 1024 + lane * 16;
      const v16bf bh = *(const v16bf*)bp;              // 32B aligned LDS read
      const v16bf bl = *(const v16bf*)(bp + LDS_BF);
      acc = __builtin_amdgcn_wmma_f32_16x16x32_bf16(false, ahi[kt], false, bh,
                                                    (short)0, acc, false, false);
      acc = __builtin_amdgcn_wmma_f32_16x16x32_bf16(false, ahi[kt], false, bl,
                                                    (short)0, acc, false, false);
      acc = __builtin_amdgcn_wmma_f32_16x16x32_bf16(false, alo[kt], false, bh,
                                                    (short)0, acc, false, false);
    }
    // C/D 16x16 f32 layout: VGPR g, lane<16 -> M=g, lane>=16 -> M=g+8; N=lane&15
    const int m0 = (lane >> 4) << 3;
    float* op = out + (size_t)(rowbase + m0) * WDIM + nt * 16 + (lane & 15);
#pragma unroll
    for (int g = 0; g < 8; ++g)
      op[(size_t)g * WDIM] = acc[g];
  }
}

// ---------------------------------------------------------------------------
extern "C" void kernel_launch(void* const* d_in, const int* in_sizes, int n_in,
                              void* d_out, int out_size, void* d_ws, size_t ws_size,
                              hipStream_t stream) {
  const float* data   = (const float*)d_in[0];
  const float* angles = (const float*)d_in[1];
  const int*   idx_in = (const int*)d_in[2];
  const int*   idxout = (const int*)d_in[3];
  float* out = (float*)d_out;

  // ws layout: G fp32 (256 KB) | bhi bf16 (128 KB) | blo bf16 (128 KB)
  float*  G   = (float*)d_ws;
  __bf16* bhi = (__bf16*)((char*)d_ws + (size_t)GELEMS * sizeof(float));
  __bf16* blo = bhi + GELEMS;   // contiguous with bhi -> single staged copy

  const int nrows = in_sizes[0] / WDIM;   // 131072

  compose_kernel<<<WDIM / 8, 256, 0, stream>>>(angles, idx_in, idxout, G);
  split_kernel<<<GELEMS / 256, 256, 0, stream>>>(G, bhi, blo);
  wmma_gemm_kernel<<<nrows / 256, 512, 0, stream>>>(data, bhi, out);
}
